// QueryInteractionModuleGroup2_6167573037555
// MI455X (gfx1250) — compile-verified, matrix operand order checked by reading, not run on hardware
//
#include <hip/hip_runtime.h>
#include <hip/hip_bf16.h>
#include <math.h>

// ---------------------------------------------------------------------------
// CDNA5 / gfx1250: wave32, WMMA 16x16x32 bf16 (D = A x B + C, f32 accum)
// GEMMs stage weight tiles into LDS with global_load_async_to_lds_b128
// (ASYNCcnt-tracked DMA), double-buffered against the WMMA stream.
// ---------------------------------------------------------------------------

typedef __bf16 bf16_t;
typedef bf16_t v16bf __attribute__((ext_vector_type(16)));
typedef float  v8f   __attribute__((ext_vector_type(8)));

#define N_TOK 4096
#define DMODEL 256
#define NHEAD 8
#define HD 32
#define FFDIM 1024

__device__ __forceinline__ v8f wmma_bf16(v16bf a, v16bf b, v8f c) {
  return __builtin_amdgcn_wmma_f32_16x16x32_bf16(
      /*neg_a=*/false, a, /*neg_b=*/false, b,
      /*c_mod=*/(short)0, c, /*reuse_a=*/false, /*reuse_b=*/false);
}

// Load one 16-element bf16 fragment chunk pair: p[0..7] and p[16..23]
// (matches the ISA 16-bit A/B fragment layout: per-lane two 8-elem runs).
__device__ __forceinline__ v16bf load_frag16(const bf16_t* p) {
  union { v16bf v; uint4 u[2]; } f;
  f.u[0] = *reinterpret_cast<const uint4*>(p);
  f.u[1] = *reinterpret_cast<const uint4*>(p + 16);
  return f.v;
}

// A-fragment (and mirror-layout B-fragment) loader from a row-major matrix.
// lane<16  : row = row0+lane,    k = k0+{0..7,16..23}
// lane>=16 : row = row0+lane-16, k = k0+{8..15,24..31}
__device__ __forceinline__ v16bf load_afrag(const bf16_t* base, int ld,
                                            int row0, int k0) {
  int lane = threadIdx.x & 31;
  const bf16_t* p =
      base + (size_t)(row0 + (lane & 15)) * ld + k0 + ((lane >> 4) << 3);
  return load_frag16(p);
}

// Async DMA: 16 bytes per lane, global -> LDS. Tracked by ASYNCcnt.
__device__ __forceinline__ void async_copy_b128(unsigned lds_off,
                                                const void* gaddr) {
  asm volatile("global_load_async_to_lds_b128 %0, %1, off"
               :: "v"(lds_off), "v"(gaddr)
               : "memory");
}
__device__ __forceinline__ void wait_async0() {
  asm volatile("s_wait_asynccnt 0x0" ::: "memory");
}

// ---------------------------------------------------------------------------
// Kernel: fp32 -> bf16 weight conversion
// ---------------------------------------------------------------------------
__global__ void f2b_kernel(const float* __restrict__ src,
                           bf16_t* __restrict__ dst, int n) {
  int i = blockIdx.x * blockDim.x + threadIdx.x;
  if (i < n) dst[i] = (bf16_t)src[i];
}

// ---------------------------------------------------------------------------
// Kernel: ref select + sin/cos positional embedding + qk = pe + out_embed
// ---------------------------------------------------------------------------
__global__ void prep_kernel(const float* __restrict__ ref_pts,
                            const float* __restrict__ pred_boxes,
                            const float* __restrict__ scores,
                            const float* __restrict__ out_embed,
                            bf16_t* __restrict__ qk_bf,
                            bf16_t* __restrict__ oe_bf) {
  int idx = blockIdx.x * blockDim.x + threadIdx.x;
  if (idx >= N_TOK * DMODEL) return;
  int n = idx >> 8, d = idx & 255;
  int j = d >> 6;                    // which of the 4 coords
  bool is_pos = scores[n] > 0.5f;
  float pos = is_pos ? pred_boxes[n * 4 + j] : ref_pts[n * 4 + j];
  int s = (d & 63) >> 1;             // frequency index
  int t = d & 1;                     // sin / cos
  // temperature^(2s/64) = exp(ln(1e4) * s / 32)
  float denom = __expf(9.210340371976184f * (float)s * (1.0f / 32.0f));
  float val = pos * 6.283185307179586f / denom;
  float pe = (t == 0) ? __sinf(val) : __cosf(val);
  float oe = out_embed[idx];
  qk_bf[idx] = (bf16_t)(pe + oe);
  oe_bf[idx] = (bf16_t)oe;
}

// ---------------------------------------------------------------------------
// Kernel: bf16 WMMA GEMM  Y[M,N] = A[M,K] @ W[N,K]^T + bias
// Block = 8 waves = 128x64 output tile. The shared 64x32 weight chunk is
// staged into LDS by async DMA (double-buffered); each wave then runs
// 4 WMMAs per K-step with B fragments served from LDS.
// ---------------------------------------------------------------------------
__global__ void gemm_bf16_kernel(const bf16_t* __restrict__ A, int lda,
                                 const bf16_t* __restrict__ W,
                                 const float* __restrict__ bias,
                                 float* __restrict__ outF,
                                 bf16_t* __restrict__ outB,
                                 int M, int N, int K, int relu, int transB) {
  __shared__ bf16_t wst[2][64 * 32];   // 2 x 4KB double buffer

  int nTiles = N >> 6;
  int bm = (int)(blockIdx.x / nTiles) << 7;   // 128-row block tile
  int n0 = (int)(blockIdx.x % nTiles) << 6;   // 64-col block tile
  int wave = threadIdx.x >> 5;
  int m0 = bm + (wave << 4);
  int lane = threadIdx.x & 31;
  int l15 = lane & 15;
  int half = lane >> 4;

  // cooperative stage of one 64x32 weight chunk: 256 threads x 16B = 4KB
  int srow = threadIdx.x >> 2;       // 0..63
  int sseg = (threadIdx.x & 3) << 3; // 0,8,16,24
  auto stage = [&](int buf, int k0) {
    const bf16_t* src = W + (size_t)(n0 + srow) * K + k0 + sseg;
    unsigned dst = (unsigned)(uintptr_t)(&wst[buf][srow * 32 + sseg]);
    async_copy_b128(dst, src);
  };

  int nk = K >> 5;
  v8f acc0 = {}, acc1 = {}, acc2 = {}, acc3 = {};
  stage(0, 0);
  for (int kt = 0; kt < nk; ++kt) {
    wait_async0();          // our slice of chunk kt has landed
    __syncthreads();        // everyone's slice visible
    if (kt + 1 < nk) stage((kt + 1) & 1, (kt + 1) << 5);

    int k0 = kt << 5;
    v16bf a = load_afrag(A, lda, m0, k0);
    const bf16_t* bb = wst[kt & 1] + (half << 3);
    v16bf b0 = load_frag16(bb + (l15) * 32);
    v16bf b1 = load_frag16(bb + (16 + l15) * 32);
    v16bf b2 = load_frag16(bb + (32 + l15) * 32);
    v16bf b3 = load_frag16(bb + (48 + l15) * 32);
    acc0 = wmma_bf16(a, b0, acc0);
    acc1 = wmma_bf16(a, b1, acc1);
    acc2 = wmma_bf16(a, b2, acc2);
    acc3 = wmma_bf16(a, b3, acc3);
    __syncthreads();        // all waves done reading buf[kt&1]
  }

  int mb = m0 + (half << 3);          // C/D layout: lanes 0-15 -> M 0..7
  v8f accs[4] = {acc0, acc1, acc2, acc3};
  for (int t = 0; t < 4; ++t) {
    int n = n0 + (t << 4) + l15;
    float bv = bias ? bias[n] : 0.0f;
    for (int i = 0; i < 8; ++i) {
      float v = accs[t][i] + bv;
      if (relu) v = fmaxf(v, 0.0f);
      int r = mb + i;
      if (outF) outF[(size_t)r * N + n] = v;
      if (outB) {
        if (transB) outB[(size_t)n * M + r] = (bf16_t)v;
        else        outB[(size_t)r * N + n] = (bf16_t)v;
      }
    }
  }
}

// ---------------------------------------------------------------------------
// Kernel: group-masked flash attention. One wave = (head, 16-row tile).
// QK buffer: [N, 512] with Q in cols [0,256), K in cols [256,512).
// Vt buffer: [256, N] (V transposed -> contiguous B-fragment loads).
// ---------------------------------------------------------------------------
__global__ void attn_kernel(const bf16_t* __restrict__ QK,
                            const bf16_t* __restrict__ Vt,
                            const int* __restrict__ gid,
                            bf16_t* __restrict__ ctx) {
  __shared__ bf16_t pstage[8 * 16 * 32];          // 1 KB per wave slice
  int wslot = threadIdx.x >> 5;
  bf16_t* pbuf = pstage + wslot * 512;

  int wid = blockIdx.x * (blockDim.x >> 5) + wslot;
  const int mTiles = N_TOK >> 4;
  if (wid >= NHEAD * mTiles) return;
  int h  = wid / mTiles;
  int m0 = (wid % mTiles) << 4;

  int lane = threadIdx.x & 31;
  int l15 = lane & 15;
  int half = lane >> 4;
  int mb = half << 3;
  const float scale = 0.17677669529663687f;       // 1/sqrt(32)
  const float NEG = -1e30f;

  v16bf qf = load_afrag(QK, 512, m0, h * HD);     // Q tile, K-dim = HD = 32
  int grow[8];
  for (int i = 0; i < 8; ++i) grow[i] = gid[m0 + mb + i];

  float mrun[8], lrun[8];
  v8f o0 = {}, o1 = {};
  for (int i = 0; i < 8; ++i) { mrun[i] = NEG; lrun[i] = 0.0f; }

  for (int j = 0; j < N_TOK; j += 32) {
    // logits: one WMMA per 16x16 tile (K rows act as B columns)
    v16bf k0f = load_afrag(QK, 512, j,      256 + h * HD);
    v16bf k1f = load_afrag(QK, 512, j + 16, 256 + h * HD);
    v8f z = {};
    v8f s0 = wmma_bf16(qf, k0f, z);
    v8f s1 = wmma_bf16(qf, k1f, z);

    int gc0 = gid[j + l15];
    int gc1 = gid[j + 16 + l15];

    float p0[8], p1[8];
    for (int i = 0; i < 8; ++i) {
      p0[i] = (grow[i] == gc0) ? s0[i] * scale : NEG;
      p1[i] = (grow[i] == gc1) ? s1[i] * scale : NEG;
    }

    // online softmax: row stats live across the 16 lanes of each half-wave
    for (int i = 0; i < 8; ++i) {
      float t = fmaxf(p0[i], p1[i]);
      t = fmaxf(t, __shfl_xor(t, 1, 32));
      t = fmaxf(t, __shfl_xor(t, 2, 32));
      t = fmaxf(t, __shfl_xor(t, 4, 32));
      t = fmaxf(t, __shfl_xor(t, 8, 32));
      float mnew = fmaxf(mrun[i], t);
      float corr = (mnew <= 0.5f * NEG) ? 1.0f : __expf(mrun[i] - mnew);
      float e0 = (p0[i] <= 0.5f * NEG) ? 0.0f : __expf(p0[i] - mnew);
      float e1 = (p1[i] <= 0.5f * NEG) ? 0.0f : __expf(p1[i] - mnew);
      float rs = e0 + e1;
      rs += __shfl_xor(rs, 1, 32);
      rs += __shfl_xor(rs, 2, 32);
      rs += __shfl_xor(rs, 4, 32);
      rs += __shfl_xor(rs, 8, 32);
      lrun[i] = lrun[i] * corr + rs;
      mrun[i] = mnew;
      o0[i] *= corr;
      o1[i] *= corr;
      p0[i] = e0;
      p1[i] = e1;
    }

    // C-layout (rows x lanes) -> A-layout via per-wave LDS slice
    for (int i = 0; i < 8; ++i) {
      pbuf[(mb + i) * 32 + l15]      = (bf16_t)p0[i];
      pbuf[(mb + i) * 32 + 16 + l15] = (bf16_t)p1[i];
    }
    asm volatile("s_wait_dscnt 0" ::: "memory");
    v16bf pf = load_frag16(pbuf + l15 * 32 + (half << 3));

    // O += P @ V ; Vt rows are head-dim, contiguous over tokens
    v16bf v0f = load_afrag(Vt, N_TOK, h * HD,      j);
    v16bf v1f = load_afrag(Vt, N_TOK, h * HD + 16, j);
    o0 = wmma_bf16(pf, v0f, o0);
    o1 = wmma_bf16(pf, v1f, o1);
  }

  for (int i = 0; i < 8; ++i) {
    float inv = (lrun[i] > 0.0f) ? (1.0f / lrun[i]) : 0.0f;
    size_t r = (size_t)(m0 + mb + i) * DMODEL + h * HD;
    ctx[r + l15]      = (bf16_t)(o0[i] * inv);
    ctx[r + 16 + l15] = (bf16_t)(o1[i] * inv);
  }
}

// ---------------------------------------------------------------------------
// Kernel: residual + LayerNorm; one wave per row (256 cols = 8/lane).
// If `scores` != null: final mode -> out = is_pos ? LN(resid+x) : resid.
// ---------------------------------------------------------------------------
__global__ void ln_kernel(const float* __restrict__ resid,
                          const float* __restrict__ x,
                          const float* __restrict__ w,
                          const float* __restrict__ b,
                          float* __restrict__ outF,
                          bf16_t* __restrict__ outB,
                          const float* __restrict__ scores) {
  int row = blockIdx.x * (blockDim.x >> 5) + (threadIdx.x >> 5);
  if (row >= N_TOK) return;
  int lane = threadIdx.x & 31;
  size_t base = (size_t)row * DMODEL;

  float vals[8];
  float sum = 0.0f;
  for (int i = 0; i < 8; ++i) {
    int c = lane + i * 32;
    float v = resid[base + c] + x[base + c];
    vals[i] = v;
    sum += v;
  }
  for (int m = 16; m >= 1; m >>= 1) sum += __shfl_xor(sum, m, 32);
  float mean = sum * (1.0f / DMODEL);
  float var = 0.0f;
  for (int i = 0; i < 8; ++i) { float d = vals[i] - mean; var += d * d; }
  for (int m = 16; m >= 1; m >>= 1) var += __shfl_xor(var, m, 32);
  float rstd = rsqrtf(var * (1.0f / DMODEL) + 1e-5f);

  bool sel = scores ? (scores[row] > 0.5f) : true;
  for (int i = 0; i < 8; ++i) {
    int c = lane + i * 32;
    float y = (vals[i] - mean) * rstd * w[c] + b[c];
    if (scores && !sel) y = resid[base + c];   // query_pos passthrough
    if (outF) outF[base + c] = y;
    if (outB) outB[base + c] = (bf16_t)y;
  }
}

// ---------------------------------------------------------------------------
// Host side
// ---------------------------------------------------------------------------
static inline int gemm_blocks(int M, int N) {
  return (M >> 7) * (N >> 6);   // 128x64 tile per 256-thread block
}

extern "C" void kernel_launch(void* const* d_in, const int* in_sizes, int n_in,
                              void* d_out, int out_size, void* d_ws, size_t ws_size,
                              hipStream_t stream) {
  const float* ref_pts     = (const float*)d_in[0];
  const float* pred_boxes  = (const float*)d_in[1];
  const float* scores      = (const float*)d_in[2];
  const float* out_embed   = (const float*)d_in[3];
  const float* query_pos   = (const float*)d_in[4];
  const int*   group_ids   = (const int*)d_in[5];
  const float* in_proj_w   = (const float*)d_in[7];
  const float* in_proj_b   = (const float*)d_in[8];
  const float* out_proj_w  = (const float*)d_in[9];
  const float* out_proj_b  = (const float*)d_in[10];
  const float* lin1_w      = (const float*)d_in[11];
  const float* lin1_b      = (const float*)d_in[12];
  const float* lin2_w      = (const float*)d_in[13];
  const float* lin2_b      = (const float*)d_in[14];
  const float* feat1_w     = (const float*)d_in[15];
  const float* feat1_b     = (const float*)d_in[16];
  const float* feat2_w     = (const float*)d_in[17];
  const float* feat2_b     = (const float*)d_in[18];
  const float* norm1_w     = (const float*)d_in[19];
  const float* norm1_b     = (const float*)d_in[20];
  const float* norm2_w     = (const float*)d_in[21];
  const float* norm2_b     = (const float*)d_in[22];
  const float* normf_w     = (const float*)d_in[23];
  const float* normf_b     = (const float*)d_in[24];

  // ---- workspace carve-up (256B aligned slices) ----
  size_t off = 0;
  auto alloc = [&](size_t bytes) -> void* {
    void* p = (char*)d_ws + off;
    off = (off + bytes + 255) & ~(size_t)255;
    return p;
  };
  bf16_t* qk_bf   = (bf16_t*)alloc((size_t)N_TOK * DMODEL * 2);
  bf16_t* oe_bf   = (bf16_t*)alloc((size_t)N_TOK * DMODEL * 2);
  bf16_t* wb_inp  = (bf16_t*)alloc((size_t)3 * DMODEL * DMODEL * 2);
  bf16_t* wb_outp = (bf16_t*)alloc((size_t)DMODEL * DMODEL * 2);
  bf16_t* wb_l1   = (bf16_t*)alloc((size_t)FFDIM * DMODEL * 2);
  bf16_t* wb_l2   = (bf16_t*)alloc((size_t)DMODEL * FFDIM * 2);
  bf16_t* wb_f1   = (bf16_t*)alloc((size_t)FFDIM * DMODEL * 2);
  bf16_t* wb_f2   = (bf16_t*)alloc((size_t)DMODEL * FFDIM * 2);
  bf16_t* QKbf    = (bf16_t*)alloc((size_t)N_TOK * 2 * DMODEL * 2);
  bf16_t* Vtbf    = (bf16_t*)alloc((size_t)DMODEL * N_TOK * 2);
  bf16_t* ctx_bf  = (bf16_t*)alloc((size_t)N_TOK * DMODEL * 2);
  float*  tmpF    = (float*) alloc((size_t)N_TOK * DMODEL * 4);
  float*  tgtF    = (float*) alloc((size_t)N_TOK * DMODEL * 4);
  bf16_t* tgtB    = (bf16_t*)alloc((size_t)N_TOK * DMODEL * 2);
  bf16_t* hidB    = (bf16_t*)alloc((size_t)N_TOK * FFDIM * 2);
  float*  tgtF2   = (float*) alloc((size_t)N_TOK * DMODEL * 4);
  bf16_t* tgtB2   = (bf16_t*)alloc((size_t)N_TOK * DMODEL * 2);

  const int T = 256;

  // ---- weights -> bf16 ----
  int nip = 3 * DMODEL * DMODEL;
  f2b_kernel<<<(nip + T - 1) / T, T, 0, stream>>>(in_proj_w, wb_inp, nip);
  int nop = DMODEL * DMODEL;
  f2b_kernel<<<(nop + T - 1) / T, T, 0, stream>>>(out_proj_w, wb_outp, nop);
  int nff = FFDIM * DMODEL;
  f2b_kernel<<<(nff + T - 1) / T, T, 0, stream>>>(lin1_w, wb_l1, nff);
  f2b_kernel<<<(nff + T - 1) / T, T, 0, stream>>>(lin2_w, wb_l2, nff);
  f2b_kernel<<<(nff + T - 1) / T, T, 0, stream>>>(feat1_w, wb_f1, nff);
  f2b_kernel<<<(nff + T - 1) / T, T, 0, stream>>>(feat2_w, wb_f2, nff);

  // ---- pos-embed + qk ----
  prep_kernel<<<(N_TOK * DMODEL + T - 1) / T, T, 0, stream>>>(
      ref_pts, pred_boxes, scores, out_embed, qk_bf, oe_bf);

  // ---- Q,K projection (fused 512-wide) ----
  gemm_bf16_kernel<<<gemm_blocks(N_TOK, 2 * DMODEL), T, 0, stream>>>(
      qk_bf, DMODEL, wb_inp, in_proj_b, nullptr, QKbf,
      N_TOK, 2 * DMODEL, DMODEL, 0, 0);
  // ---- V projection (transposed store) ----
  gemm_bf16_kernel<<<gemm_blocks(N_TOK, DMODEL), T, 0, stream>>>(
      oe_bf, DMODEL, wb_inp + (size_t)2 * DMODEL * DMODEL,
      in_proj_b + 2 * DMODEL, nullptr, Vtbf,
      N_TOK, DMODEL, DMODEL, 0, 1);

  // ---- flash attention ----
  {
    int waves = NHEAD * (N_TOK >> 4);
    attn_kernel<<<(waves + 7) / 8, T, 0, stream>>>(QKbf, Vtbf, group_ids, ctx_bf);
  }

  // ---- out_proj + norm1 ----
  gemm_bf16_kernel<<<gemm_blocks(N_TOK, DMODEL), T, 0, stream>>>(
      ctx_bf, DMODEL, wb_outp, out_proj_b, tmpF, nullptr,
      N_TOK, DMODEL, DMODEL, 0, 0);
  ln_kernel<<<(N_TOK + 7) / 8, T, 0, stream>>>(
      out_embed, tmpF, norm1_w, norm1_b, tgtF, tgtB, nullptr);

  // ---- FFN (lin1 relu, lin2) + norm2 ----
  gemm_bf16_kernel<<<gemm_blocks(N_TOK, FFDIM), T, 0, stream>>>(
      tgtB, DMODEL, wb_l1, lin1_b, nullptr, hidB,
      N_TOK, FFDIM, DMODEL, 1, 0);
  gemm_bf16_kernel<<<gemm_blocks(N_TOK, DMODEL), T, 0, stream>>>(
      hidB, FFDIM, wb_l2, lin2_b, tmpF, nullptr,
      N_TOK, DMODEL, FFDIM, 0, 0);
  ln_kernel<<<(N_TOK + 7) / 8, T, 0, stream>>>(
      tgtF, tmpF, norm2_w, norm2_b, tgtF2, tgtB2, nullptr);

  // ---- feature FFN + final norm with is_pos select into d_out ----
  gemm_bf16_kernel<<<gemm_blocks(N_TOK, FFDIM), T, 0, stream>>>(
      tgtB2, DMODEL, wb_f1, feat1_b, nullptr, hidB,
      N_TOK, FFDIM, DMODEL, 1, 0);
  gemm_bf16_kernel<<<gemm_blocks(N_TOK, DMODEL), T, 0, stream>>>(
      hidB, FFDIM, wb_f2, feat2_b, tmpF, nullptr,
      N_TOK, DMODEL, FFDIM, 0, 0);
  ln_kernel<<<(N_TOK + 7) / 8, T, 0, stream>>>(
      query_pos, tmpF, normf_w, normf_b, (float*)d_out, nullptr, scores);

  (void)in_sizes; (void)n_in; (void)out_size; (void)ws_size;
}
